// HGT_71846212928266
// MI455X (gfx1250) — matrix operand order repaired and checked by reading, not compiled, not run on hardware
//
#include <hip/hip_runtime.h>
#include <math.h>

// HGT forward for MI455X (gfx1250, wave32).
// Dense GEMMs: v_wmma_f32_16x16x32_f16 (fp32 accumulate), fp16 intermediates,
// double-buffered async global->LDS staging (global_load_async_to_lds_b128,
// s_wait_asynccnt) overlapping DMA with WMMA; one barrier per K-step.
// Edge phase: L2-resident gathers + float atomics for segment softmax/scatter.

#define NN 50000
#define EE 200000
#define CC 512
#define HH 8
#define DD 64

typedef __attribute__((ext_vector_type(16))) _Float16 v16h;
typedef __attribute__((ext_vector_type(8)))  _Float16 v8h;
typedef __attribute__((ext_vector_type(4)))  _Float16 v4h;
typedef __attribute__((ext_vector_type(8)))  float    v8f;

#define BM 128
#define BN 64
#define BK 32
#define LDAS 40   // padded LDS row stride (halfs) for A tile [row][k]
#define LDBT 40   // padded LDS row stride (halfs) for B tile [col][k] (transposed)
#define ASZ (BM * LDAS)   // one A buffer (halfs)
#define BSZ (BN * LDBT)   // one B buffer (halfs)

__device__ __forceinline__ float gelu_exact(float x) {
  return 0.5f * x * (1.0f + erff(x * 0.70710678118654752f));
}

// ---------------------------------------------------------------------------
// Double-buffered GEMM main loop. A fp16 (async DMA to LDS in FULL path),
// B fp32 weights converted to fp16 while staged transposed.
// ---------------------------------------------------------------------------
template <bool FULL>
__device__ __forceinline__ void gemm_mainloop(
    const _Float16* __restrict__ A, const float* __restrict__ B,
    _Float16* As, _Float16* Bt,
    int tid, int m0, int n0, int M, int K, int lda, int ldb,
    int wm, int wn, int lrow, int lhi, int bn, int bk, v8f (&acc)[2][2])
{
  auto stageA = [&](int k0, _Float16* dstAs) {
#pragma unroll
    for (int i = 0; i < 2; ++i) {
      int q = tid + (i << 8);        // 0..511 groups of 8 halfs
      int r = q >> 2;                // row 0..127
      int c = (q & 3) << 3;          // k 0,8,16,24
      const _Float16* p = A + (long)(m0 + r) * lda + (k0 + c);
      if constexpr (FULL) {
        unsigned long long ga = (unsigned long long)(size_t)p;
        unsigned dsa = (unsigned)(size_t)(void*)&dstAs[r * LDAS + c];
        // CDNA5 async DMA: LDS[dsa] <- MEM[ga], tracked by ASYNCcnt
        asm volatile("global_load_async_to_lds_b128 %0, %1, off"
                     :: "v"(dsa), "v"(ga) : "memory");
        if (k0 + 2 * BK < K) __builtin_prefetch(p + 2 * BK, 0, 1);  // global_prefetch_b8
      } else {
        v8h v = {};
        if ((m0 + r) < M) v = *(const v8h*)p;
        *(v8h*)&dstAs[r * LDAS + c] = v;
      }
    }
  };
  auto stageB = [&](int k0, _Float16* dstBt) {
    const float* bp = B + (long)(k0 + bk) * ldb + (n0 + bn);
    v8h t;
#pragma unroll
    for (int j = 0; j < 8; ++j) t[j] = (_Float16)bp[(long)j * ldb];
    *(v8h*)&dstBt[bn * LDBT + bk] = t;
  };

  // prologue: stage tile 0 into buffer 0
  stageA(0, As);
  stageB(0, Bt);
  if constexpr (FULL) asm volatile("s_wait_asynccnt 0x0" ::: "memory");
  __syncthreads();

  const int nk = K / BK;              // K is always a multiple of 32
  for (int t = 0; t < nk; ++t) {
    _Float16* Ac = As + (t & 1) * ASZ;
    _Float16* Bc = Bt + (t & 1) * BSZ;
    // stage tile t+1 into the idle buffer while computing tile t
    if (t + 1 < nk) {
      stageA((t + 1) * BK, As + ((t + 1) & 1) * ASZ);
      stageB((t + 1) * BK, Bt + ((t + 1) & 1) * BSZ);
    }

    // ---- fragments (ISA 7.12.2 layouts): all 2x ds_load_b128 ----
    v16h af[2], bf[2];
#pragma unroll
    for (int mi = 0; mi < 2; ++mi) {
      const _Float16* ap = &Ac[(wm * 32 + mi * 16 + lrow) * LDAS + (lhi << 3)];
#pragma unroll
      for (int e = 0; e < 16; ++e)
        af[mi][e] = ap[(e & 7) + ((e >> 3) << 4)];   // runs [0..7],[16..23]
    }
#pragma unroll
    for (int ni = 0; ni < 2; ++ni) {
      const _Float16* bp2 = &Bc[(wn * 32 + ni * 16 + lrow) * LDBT + (lhi << 3)];
#pragma unroll
      for (int e = 0; e < 16; ++e)
        bf[ni][e] = bp2[(e & 7) + ((e >> 3) << 4)];  // runs [0..7],[16..23]
    }
#pragma unroll
    for (int mi = 0; mi < 2; ++mi)
#pragma unroll
      for (int ni = 0; ni < 2; ++ni)
        acc[mi][ni] = __builtin_amdgcn_wmma_f32_16x16x32_f16(
            false, af[mi], false, bf[ni], (short)0, acc[mi][ni], false, false);

    // one barrier per K-step: next buffer staged + this buffer fully read
    if (t + 1 < nk) {
      if constexpr (FULL) asm volatile("s_wait_asynccnt 0x0" ::: "memory");
      __syncthreads();
    }
  }
}

// ---------------------------------------------------------------------------
// WMMA GEMM: C = A @ B (+bias); optional sigmoid-skip gate epilogue.
// OUT_H: write C as fp16 (intermediates) or fp32 (final output).
// Batched over blockIdx.z with element strides sA/sB/sC.
// 256 threads = 8 waves; tile 128(M) x 64(N); K-step 32; waves 4(M) x 2(N).
// ---------------------------------------------------------------------------
template <bool GATED, bool OUT_H>
__global__ __launch_bounds__(256) void wmma_gemm(
    const _Float16* __restrict__ Abase, const float* __restrict__ Bbase,
    const float* __restrict__ bias, const float* __restrict__ Xres,
    const float* __restrict__ skipPtr, void* __restrict__ CbaseV,
    int M, int Ncols, int K, int lda, int ldb, int ldc,
    long sA, long sB, long sC)
{
  const _Float16* A = Abase + (long)blockIdx.z * sA;
  const float*    B = Bbase + (long)blockIdx.z * sB;

  __shared__ _Float16 As[2 * ASZ];   // double-buffered
  __shared__ _Float16 Bt[2 * BSZ];

  const int tid  = threadIdx.x;
  const int lane = tid & 31;
  const int wave = tid >> 5;
  const int wm   = wave >> 1;      // 0..3
  const int wn   = wave & 1;       // 0..1
  const int lrow = lane & 15;
  const int lhi  = lane >> 4;
  const int m0   = blockIdx.x * BM;
  const int n0   = blockIdx.y * BN;
  const int bn   = tid & 63;             // B staging column
  const int bk   = (tid >> 6) << 3;      // B staging k-run: 0,8,16,24

  v8f acc[2][2] = {};
  const bool fullM = (m0 + BM) <= M;     // uniform: only last M-block partial
  if (fullM)
    gemm_mainloop<true >(A, B, As, Bt, tid, m0, n0, M, K, lda, ldb,
                         wm, wn, lrow, lhi, bn, bk, acc);
  else
    gemm_mainloop<false>(A, B, As, Bt, tid, m0, n0, M, K, lda, ldb,
                         wm, wn, lrow, lhi, bn, bk, acc);

  // ------------------------------- epilogue -------------------------------
  float g = 0.f;
  if constexpr (GATED) g = 1.f / (1.f + expf(-skipPtr[0]));
  _Float16* Ch = (_Float16*)CbaseV + (long)blockIdx.z * sC;
  float*    Cf = (float*)CbaseV + (long)blockIdx.z * sC;
#pragma unroll
  for (int mi = 0; mi < 2; ++mi)
#pragma unroll
    for (int ni = 0; ni < 2; ++ni) {
      int n = n0 + wn * 32 + ni * 16 + lrow;   // Ncols always multiple of 64
      float bv = bias ? bias[n] : 0.f;
#pragma unroll
      for (int j = 0; j < 8; ++j) {
        int m = m0 + wm * 32 + mi * 16 + (lhi << 3) + j;
        if (fullM || m < M) {
          float o = acc[mi][ni][j] + bv;
          if constexpr (GATED) o = g * o + (1.f - g) * Xres[(long)m * ldc + n];
          if constexpr (OUT_H) Ch[(long)m * ldc + n] = (_Float16)o;
          else                 Cf[(long)m * ldc + n] = o;
        }
      }
    }
}

// ---------------------------------------------------------------------------
// Converts + fills
// ---------------------------------------------------------------------------
__global__ void cvt_f16_kernel(const float* __restrict__ in,
                               _Float16* __restrict__ out, long n4) {
  long i = (long)blockIdx.x * blockDim.x + threadIdx.x;
  long s = (long)gridDim.x * blockDim.x;
  for (; i < n4; i += s) {
    float4 v = ((const float4*)in)[i];
    v4h t; t[0] = (_Float16)v.x; t[1] = (_Float16)v.y;
           t[2] = (_Float16)v.z; t[3] = (_Float16)v.w;
    ((v4h*)out)[i] = t;
  }
}
__global__ void cvt_gelu_f16_kernel(const float* __restrict__ in,
                                    _Float16* __restrict__ out, long n4) {
  long i = (long)blockIdx.x * blockDim.x + threadIdx.x;
  long s = (long)gridDim.x * blockDim.x;
  for (; i < n4; i += s) {
    float4 v = ((const float4*)in)[i];
    v4h t; t[0] = (_Float16)gelu_exact(v.x); t[1] = (_Float16)gelu_exact(v.y);
           t[2] = (_Float16)gelu_exact(v.z); t[3] = (_Float16)gelu_exact(v.w);
    ((v4h*)out)[i] = t;
  }
}
__global__ void fill_f32_kernel(float* p, float v, long n) {
  long i = (long)blockIdx.x * blockDim.x + threadIdx.x;
  long s = (long)gridDim.x * blockDim.x;
  for (; i < n; i += s) p[i] = v;
}
__global__ void fill_u32_kernel(unsigned* p, unsigned v, long n) {
  long i = (long)blockIdx.x * blockDim.x + threadIdx.x;
  long s = (long)gridDim.x * blockDim.x;
  for (; i < n; i += s) p[i] = v;
}

// ---------------------------------------------------------------------------
// Segment softmax over destinations (order-preserving uint atomicMax trick)
// ---------------------------------------------------------------------------
__device__ __forceinline__ unsigned f2ord(float f) {
  unsigned u = __float_as_uint(f);
  return (u & 0x80000000u) ? ~u : (u | 0x80000000u);
}
__device__ __forceinline__ float ord2f(unsigned u) {
  return (u & 0x80000000u) ? __uint_as_float(u & 0x7fffffffu) : __uint_as_float(~u);
}

__global__ void att_score_kernel(const _Float16* __restrict__ q,
                                 const _Float16* __restrict__ k2,
                                 const int* __restrict__ ei,
                                 const float* __restrict__ prel,
                                 float* __restrict__ scores,
                                 unsigned* __restrict__ mkey)
{
  long i = (long)blockIdx.x * blockDim.x + threadIdx.x;
  if (i >= (long)EE * HH) return;
  int h = (int)(i & (HH - 1));
  long e = i >> 3;
  int src = ei[e];
  int dst = ei[(long)EE + e];
  const v8h* qv = (const v8h*)(q + (long)dst * CC + h * DD);
  const v8h* kv = (const v8h*)(k2 + (long)src * CC + h * DD);
  float s = 0.f;
#pragma unroll
  for (int c = 0; c < 8; ++c) {
    v8h a = qv[c], b = kv[c];
#pragma unroll
    for (int j = 0; j < 8; ++j) s += (float)a[j] * (float)b[j];
  }
  s *= prel[h] * 0.125f;  // 1/sqrt(64)
  scores[i] = s;
  atomicMax(&mkey[(long)dst * HH + h], f2ord(s));
}

__global__ void att_expsum_kernel(const int* __restrict__ ei,
                                  const unsigned* __restrict__ mkey,
                                  float* __restrict__ scores,
                                  float* __restrict__ ssum)
{
  long i = (long)blockIdx.x * blockDim.x + threadIdx.x;
  if (i >= (long)EE * HH) return;
  int h = (int)(i & (HH - 1));
  long e = i >> 3;
  int dst = ei[(long)EE + e];
  float m = ord2f(mkey[(long)dst * HH + h]);
  float w = expf(scores[i] - m);
  scores[i] = w;
  atomicAdd(&ssum[(long)dst * HH + h], w);
}

__global__ void att_scatter_kernel(const int* __restrict__ ei,
                                   const float* __restrict__ scores,
                                   const float* __restrict__ ssum,
                                   const _Float16* __restrict__ v2,
                                   float* __restrict__ agg)
{
  long i = (long)blockIdx.x * blockDim.x + threadIdx.x;
  if (i >= (long)EE * HH * DD) return;
  int d = (int)(i & (DD - 1));
  int h = (int)((i >> 6) & (HH - 1));
  long e = i >> 9;
  int src = ei[e];
  int dst = ei[(long)EE + e];
  float w = scores[e * HH + h] / (ssum[(long)dst * HH + h] + 1e-16f);
  atomicAdd(&agg[(long)dst * CC + h * DD + d],
            w * (float)v2[(long)src * CC + h * DD + d]);
}

// ---------------------------------------------------------------------------
// Launch
// ---------------------------------------------------------------------------
extern "C" void kernel_launch(void* const* d_in, const int* in_sizes, int n_in,
                              void* d_out, int out_size, void* d_ws, size_t ws_size,
                              hipStream_t stream)
{
  const float* x_A  = (const float*)d_in[0];
  const float* x_B  = (const float*)d_in[1];
  const float* kW_A = (const float*)d_in[2];  const float* kb_A = (const float*)d_in[3];
  const float* qW_A = (const float*)d_in[4];  const float* qb_A = (const float*)d_in[5];
  const float* vW_A = (const float*)d_in[6];  const float* vb_A = (const float*)d_in[7];
  const float* oW_A = (const float*)d_in[8];  const float* ob_A = (const float*)d_in[9];
  const float* skip_A = (const float*)d_in[10];
  const float* kW_B = (const float*)d_in[11]; const float* kb_B = (const float*)d_in[12];
  const float* qW_B = (const float*)d_in[13]; const float* qb_B = (const float*)d_in[14];
  const float* vW_B = (const float*)d_in[15]; const float* vb_B = (const float*)d_in[16];
  const float* oW_B = (const float*)d_in[17]; const float* ob_B = (const float*)d_in[18];
  const float* skip_B = (const float*)d_in[19];
  const float* arel[3] = {(const float*)d_in[20], (const float*)d_in[23], (const float*)d_in[26]};
  const float* mrel[3] = {(const float*)d_in[21], (const float*)d_in[24], (const float*)d_in[27]};
  const float* prel[3] = {(const float*)d_in[22], (const float*)d_in[25], (const float*)d_in[28]};
  const float* linW = (const float*)d_in[29];
  const float* linb = (const float*)d_in[30];
  const int* ei[3] = {(const int*)d_in[31], (const int*)d_in[32], (const int*)d_in[33]};

  // Workspace layout (fp16 unless noted): ~724 MB total.
  size_t NC = (size_t)NN * CC;
  _Float16* kA  = (_Float16*)d_ws;
  _Float16* qA  = kA + NC;  _Float16* vA = qA + NC;
  _Float16* kB  = vA + NC;  _Float16* qB = kB + NC;  _Float16* vB = qB + NC;
  _Float16* k2  = vB + NC;  _Float16* v2 = k2 + NC;
  _Float16* xhA = v2 + NC;  _Float16* xhB = xhA + NC;
  float* aggA = (float*)(xhB + NC);
  float* aggB = aggA + NC;
  float* scores = aggB + NC;                               // E*H fp32
  unsigned* mkey = (unsigned*)(scores + (size_t)EE * HH);  // N*H
  float* ssum = (float*)(mkey + (size_t)NN * HH);          // N*H
  _Float16* ghA = k2;  // gelu(aggA) fp16, reuses k2 after edge phase
  _Float16* ghB = v2;  // gelu(aggB) fp16, reuses v2 after edge phase

  dim3 blk(256);
  const int mt = (NN + BM - 1) / BM;  // 391 M-tiles

  // ---- x -> fp16 once (51 MB each: L2-resident for the 3x re-reads) ----
  cvt_f16_kernel<<<2048, 256, 0, stream>>>(x_A, xhA, (long)(NC / 4));
  cvt_f16_kernel<<<2048, 256, 0, stream>>>(x_B, xhB, (long)(NC / 4));

  // ---- K/Q/V projections: fp16 A (async-staged), fp16 out ----
  auto proj = [&](const _Float16* X, const float* W, const float* b, _Float16* Cout) {
    wmma_gemm<false, true><<<dim3(mt, CC / BN, 1), blk, 0, stream>>>(
        X, W, b, nullptr, nullptr, Cout, NN, CC, CC, CC, CC, CC, 0, 0, 0);
  };
  proj(xhA, kW_A, kb_A, kA); proj(xhA, qW_A, qb_A, qA); proj(xhA, vW_A, vb_A, vA);
  proj(xhB, kW_B, kb_B, kB); proj(xhB, qW_B, qb_B, qB); proj(xhB, vW_B, vb_B, vB);

  fill_f32_kernel<<<2048, 256, 0, stream>>>(aggA, 0.f, (long)NC);
  fill_f32_kernel<<<2048, 256, 0, stream>>>(aggB, 0.f, (long)NC);

  // relations: r1 A->B (q_B,k_A,v_A), r2 B->A, r3 A->A; r2+r3 sum into aggA
  const _Float16* relQ[3]  = {qB, qA, qA};
  const _Float16* relK[3]  = {kA, kB, kA};
  const _Float16* relV[3]  = {vA, vB, vA};
  float*          relAg[3] = {aggB, aggA, aggA};

  unsigned ehBlocks  = (unsigned)(((long)EE * HH + 255) / 256);
  unsigned ehdBlocks = (unsigned)(((long)EE * HH * DD + 255) / 256);

  for (int r = 0; r < 3; ++r) {
    // per-head relation transforms (fp16 in/out), batched grid.z = H
    wmma_gemm<false, true><<<dim3(mt, 1, HH), blk, 0, stream>>>(
        relK[r], arel[r], nullptr, nullptr, nullptr, k2,
        NN, DD, DD, CC, DD, CC, (long)DD, (long)DD * DD, (long)DD);
    wmma_gemm<false, true><<<dim3(mt, 1, HH), blk, 0, stream>>>(
        relV[r], mrel[r], nullptr, nullptr, nullptr, v2,
        NN, DD, DD, CC, DD, CC, (long)DD, (long)DD * DD, (long)DD);

    // segment softmax state: mkey = f2ord(-FLT_MAX) = 0x00800000, ssum = 0
    fill_u32_kernel<<<512, 256, 0, stream>>>(mkey, 0x00800000u, (long)NN * HH);
    fill_f32_kernel<<<512, 256, 0, stream>>>(ssum, 0.f, (long)NN * HH);

    att_score_kernel<<<ehBlocks, blk, 0, stream>>>(relQ[r], k2, ei[r], prel[r], scores, mkey);
    att_expsum_kernel<<<ehBlocks, blk, 0, stream>>>(ei[r], mkey, scores, ssum);
    att_scatter_kernel<<<ehdBlocks, blk, 0, stream>>>(ei[r], scores, ssum, v2, relAg[r]);
  }

  // ---- out_fn: gh = fp16(gelu(agg)); h = g*(gh@oW + ob) + (1-g)*x (fp16, reuse kA/kB)
  cvt_gelu_f16_kernel<<<2048, 256, 0, stream>>>(aggA, ghA, (long)(NC / 4));
  cvt_gelu_f16_kernel<<<2048, 256, 0, stream>>>(aggB, ghB, (long)(NC / 4));
  wmma_gemm<true, true><<<dim3(mt, CC / BN, 1), blk, 0, stream>>>(
      ghA, oW_A, ob_A, x_A, skip_A, kA, NN, CC, CC, CC, CC, CC, 0, 0, 0);
  wmma_gemm<true, true><<<dim3(mt, CC / BN, 1), blk, 0, stream>>>(
      ghB, oW_B, ob_B, x_B, skip_B, kB, NN, CC, CC, CC, CC, CC, 0, 0, 0);

  // ---- trailing Linear(512,128): fp16 in, fp32 out; concat via row offset ----
  float* out = (float*)d_out;
  wmma_gemm<false, false><<<dim3(mt, 128 / BN, 1), blk, 0, stream>>>(
      kA, linW, linb, nullptr, nullptr, out, NN, 128, CC, CC, 128, 128, 0, 0, 0);
  wmma_gemm<false, false><<<dim3(mt, 128 / BN, 1), blk, 0, stream>>>(
      kB, linW, linb, nullptr, nullptr, out + (size_t)NN * 128, NN, 128, CC, CC, 128, 128, 0, 0, 0);
}
